// HM_LSTM_45878840656513
// MI455X (gfx1250) — matrix operand order repaired and checked by reading, not compiled, not run on hardware
//
#include <hip/hip_runtime.h>
#include <hip/hip_bf16.h>

#define Hh 512
#define Bb 256
#define Tt 512
#define Dd 64
#define NPAD 2112                    // 4H+1 = 2049 padded up to 33*64
#define NBLK 66                      // 33 n-strips * 2 M-halves
#define NTHR 256
#define KSLABS (Hh / 32)             // 16
#define LDS_ROW 40                   // 32 bf16 payload + 8 pad (80B, 20-bank stride)

typedef __bf16 bf16;
typedef __attribute__((ext_vector_type(16))) __bf16 v16bf;
typedef __attribute__((ext_vector_type(8)))  float  v8f;
typedef __attribute__((ext_vector_type(4)))  int    v4i;
typedef __attribute__((address_space(1))) v4i* gptr4;
typedef __attribute__((address_space(3))) v4i* lptr4;

#if defined(__has_builtin)
#if __has_builtin(__builtin_amdgcn_global_load_async_to_lds_b128)
#define USE_ASYNC_LDS 1
#endif
#endif

struct Params {
  const float* x;
  const float* bias1;
  const float* bias2;
  const float* l1b;
  const float* l2b;
  const bf16* U11_1t; const bf16* U21_1t; const bf16* W01_1t;
  const bf16* U11_2t; const bf16* W01_2t;
  const bf16* l1W;    const bf16* l2W;
  float* s1; float* s2;
  float* c1; float* c2; float* h2f;
  bf16* h1; bf16* h2; bf16* h1z; bf16* h2z;
  float* z1; float* z2;          // [2][Bb] ping-pong
  unsigned* bar;
  float* out;
};

// ---------------- fragment loaders (CDNA5 wave32 WMMA layouts) ----------------
// A (16x32 bf16): lane L holds row M=L&15; elems 0..15 = K kb..kb+15,
// kb = (L>>4)*16.
__device__ inline v16bf load_fragA(const bf16* A, int lda, int m0, int k0, int lane) {
  int r  = m0 + (lane & 15);
  int kb = k0 + ((lane >> 4) << 4);
  const bf16* p = A + (size_t)r * lda + kb;
  union { v16bf v; uint4 q[2]; } u;
  u.q[0] = *(const uint4*)p;
  u.q[1] = *(const uint4*)(p + 8);
  return u.v;
}
// A from fp32 source (x input), converted on the fly.
__device__ inline v16bf load_fragA_f32(const float* A, int lda, int m0, int k0, int lane) {
  int r  = m0 + (lane & 15);
  int kb = k0 + ((lane >> 4) << 4);
  const float* p = A + (size_t)r * lda + kb;
  v16bf v;
#pragma unroll
  for (int i = 0; i < 8; ++i) v[i] = (bf16)p[i];
#pragma unroll
  for (int i = 0; i < 8; ++i) v[8 + i] = (bf16)p[8 + i];
  return v;
}
// B (32x16 bf16) from pre-transposed [N][K] weights in global: lane L holds
// col N=L&15, 16 contiguous K values at k0 + (L>>4)*16 -> one 32B load.
__device__ inline v16bf load_fragB(const bf16* Bt, int ldb, int n0, int k0, int lane) {
  const bf16* p = Bt + (size_t)(n0 + (lane & 15)) * ldb + k0 + ((lane >> 4) << 4);
  union { v16bf v; uint4 q[2]; } u;
  u.q[0] = ((const uint4*)p)[0];
  u.q[1] = ((const uint4*)p)[1];
  return u.v;
}
// B fragment from a staged LDS slab: [64 rows][LDS_ROW bf16], 80B row stride.
__device__ inline v16bf load_fragB_lds(const bf16* lbuf, int nl0, int lane) {
  const bf16* p = lbuf + (size_t)(nl0 + (lane & 15)) * LDS_ROW + ((lane >> 4) << 4);
  union { v16bf v; uint4 q[2]; } u;
  u.q[0] = ((const uint4*)p)[0];
  u.q[1] = ((const uint4*)p)[1];
  return u.v;
}
// C/D 16x16 f32: VGPR i -> row m0 + i + 8*(lane>>4), col n0 + (lane&15)
__device__ inline void store_tile(float* S, int lds, int m0, int n0, v8f acc, int lane) {
  int col   = n0 + (lane & 15);
  int rbase = m0 + ((lane >> 4) << 3);
#pragma unroll
  for (int i = 0; i < 8; ++i) S[(size_t)(rbase + i) * lds + col] = acc[i];
}

// -------------------- async staging of one 64x32 B slab ---------------------
__device__ inline void stage_B(const bf16* Bt, int ldb, int n0, int k0, bf16* lbuf) {
  int tid = threadIdx.x;
  int row = tid >> 2;                 // 0..63
  int ch  = (tid & 3) << 3;           // element offset 0,8,16,24 (16B chunks)
  const bf16* g = Bt + (size_t)(n0 + row) * ldb + k0 + ch;
  bf16* l = lbuf + row * LDS_ROW + ch;
#ifdef USE_ASYNC_LDS
  __builtin_amdgcn_global_load_async_to_lds_b128((gptr4)g, (lptr4)l, 0, 0);
#else
  *(uint4*)l = *(const uint4*)g;
#endif
}

__device__ inline void async_fence() {
#ifdef USE_ASYNC_LDS
#if __has_builtin(__builtin_amdgcn_s_wait_asynccnt)
  __builtin_amdgcn_s_wait_asynccnt(0);
#else
  asm volatile("s_wait_asynccnt 0x0" ::: "memory");
#endif
#endif
}

// Block-cooperative GEMM over one (A, Bt) pair, K = Hh, double-buffered LDS B.
__device__ inline void block_gemm(v8f acc[4], const bf16* A, int m0,
                                  const bf16* Bt, int n0,
                                  bf16* lds0, bf16* lds1, int lane) {
  stage_B(Bt, Hh, n0, 0, lds0);
  async_fence();
  __syncthreads();
  for (int ks = 0; ks < KSLABS; ++ks) {
    bf16* cur = (ks & 1) ? lds1 : lds0;
    bf16* nxt = (ks & 1) ? lds0 : lds1;
    if (ks + 1 < KSLABS) stage_B(Bt, Hh, n0, (ks + 1) << 5, nxt);
    v16bf a = load_fragA(A, Hh, m0, ks << 5, lane);
#pragma unroll
    for (int j = 0; j < 4; ++j) {
      v16bf b = load_fragB_lds(cur, j << 4, lane);
      acc[j] = __builtin_amdgcn_wmma_f32_16x16x32_bf16(
          false, a, false, b, (short)0, acc[j], false, false);
    }
    async_fence();
    __syncthreads();
  }
}

// x-term: A is fp32 (converted on the fly), B direct from global (tiny K).
__device__ inline void gemm_accum_f32A(v8f acc[4], const float* A, int lda, int m0,
                                       const bf16* Bt, int ldb, int n0,
                                       int kslabs, int lane) {
  for (int ks = 0; ks < kslabs; ++ks) {
    int k0 = ks << 5;
    v16bf a = load_fragA_f32(A, lda, m0, k0, lane);
#pragma unroll
    for (int j = 0; j < 4; ++j) {
      v16bf b = load_fragB(Bt, ldb, n0 + (j << 4), k0, lane);
      acc[j] = __builtin_amdgcn_wmma_f32_16x16x32_bf16(
          false, a, false, b, (short)0, acc[j], false, false);
    }
  }
}

// -------------------- device-wide sense-reversal barrier --------------------
__device__ inline void grid_sync(unsigned* bar) {
  __syncthreads();
  if (threadIdx.x == 0) {
    __threadfence();
    volatile unsigned* vgen = bar + 1;
    unsigned gen = *vgen;
    unsigned prev = atomicAdd(bar, 1u);
    if (prev == NBLK - 1) {
      *(volatile unsigned*)bar = 0u;
      __threadfence();
      *vgen = gen + 1u;
    } else {
      while (*vgen == gen) __builtin_amdgcn_s_sleep(1);
    }
    __threadfence();
  }
  __syncthreads();
}

__device__ inline float sigm(float v) { return 1.f / (1.f + __expf(-v)); }
__device__ inline float hsig(float v) {
  return fminf(fmaxf((v + 1.f) * 0.5f, 0.f), 1.f);
}

// ------------------------------ persistent kernel ---------------------------
__global__ void __launch_bounds__(NTHR, 1)
hmlstm_persistent(Params p) {
  __shared__ bf16 ldsB[2][64 * LDS_ROW];   // 2 x 5KB double buffer
  const int lane = threadIdx.x & 31;
  const int wav  = threadIdx.x >> 5;
  const int gw   = blockIdx.x * (NTHR / 32) + wav;
  const int tid  = blockIdx.x * NTHR + threadIdx.x;
  // block-uniform GEMM tile: 33 n-strips x 2 M-halves
  const int n0g = (blockIdx.x >> 1) << 6;                 // 0..2048
  const int m0g = ((blockIdx.x & 1) << 7) + (wav << 4);   // 0..240

  for (int t = 0; t < Tt; ++t) {
    const int rp = t & 1, wp = rp ^ 1;

    // ---- Phase 1: s1 = h1@U11_1 + (z1*h2)@U21_1 + x_t@W01_1 -------------
    {
      v8f acc[4] = {};
      block_gemm(acc, p.h1,  m0g, p.U11_1t, n0g, ldsB[0], ldsB[1], lane);
      block_gemm(acc, p.h2z, m0g, p.U21_1t, n0g, ldsB[0], ldsB[1], lane);
      gemm_accum_f32A(acc, p.x + (size_t)t * Dd, Tt * Dd, m0g,
                      p.W01_1t, Dd, n0g, Dd / 32, lane);
#pragma unroll
      for (int j = 0; j < 4; ++j)
        store_tile(p.s1, NPAD, m0g, n0g + (j << 4), acc[j], lane);
    }
    grid_sync(p.bar);

    // ---- Phase 2: cell-1 elementwise (z_bottom == 1) ---------------------
    for (int idx = tid; idx < Bb * Hh; idx += NBLK * NTHR) {
      int b = idx >> 9, j = idx & (Hh - 1);
      const float* sr = p.s1 + (size_t)b * NPAD;
      float f  = sigm(sr[j]            + p.bias1[j]);
      float ii = sigm(sr[Hh + j]       + p.bias1[Hh + j]);
      float o  = sigm(sr[2 * Hh + j]   + p.bias1[2 * Hh + j]);
      float g  = tanhf(sr[3 * Hh + j]  + p.bias1[3 * Hh + j]);
      float zn = rintf(hsig(sr[4 * Hh] + p.bias1[4 * Hh]));
      float z  = p.z1[rp * Bb + b];
      float c  = p.c1[idx];
      float cn = ii * g + (1.f - z) * f * c;
      float hn = o * tanhf(cn);
      p.c1[idx]  = cn;
      p.h1[idx]  = (bf16)hn;
      p.h1z[idx] = (bf16)(zn * hn);
      if (j == 0) p.z1[wp * Bb + b] = zn;
    }
    grid_sync(p.bar);

    // ---- Phase 3: s2 = h2_old@U11_2 + (z1_new*h1_new)@W01_2 -------------
    {
      v8f acc[4] = {};
      block_gemm(acc, p.h2,  m0g, p.U11_2t, n0g, ldsB[0], ldsB[1], lane);
      block_gemm(acc, p.h1z, m0g, p.W01_2t, n0g, ldsB[0], ldsB[1], lane);
#pragma unroll
      for (int j = 0; j < 4; ++j)
        store_tile(p.s2, NPAD, m0g, n0g + (j << 4), acc[j], lane);
    }
    grid_sync(p.bar);

    // ---- Phase 4: cell-2 elementwise ------------------------------------
    for (int idx = tid; idx < Bb * Hh; idx += NBLK * NTHR) {
      int b = idx >> 9, j = idx & (Hh - 1);
      const float* sr = p.s2 + (size_t)b * NPAD;
      float f  = sigm(sr[j]            + p.bias2[j]);
      float ii = sigm(sr[Hh + j]       + p.bias2[Hh + j]);
      float o  = sigm(sr[2 * Hh + j]   + p.bias2[2 * Hh + j]);
      float g  = tanhf(sr[3 * Hh + j]  + p.bias2[3 * Hh + j]);
      float zn = rintf(hsig(sr[4 * Hh] + p.bias2[4 * Hh]));
      float zb = p.z1[wp * Bb + b];          // z1 updated this step
      float z  = p.z2[rp * Bb + b];
      float c  = p.c2[idx];
      float ig = ii * g;
      float cn = z * ig + (1.f - z) * (1.f - zb) * c
               + (1.f - z) * zb * (f * c + ig);
      float hp = p.h2f[idx];
      float hn = (1.f - z) * (1.f - zb) * hp
               + (z + (1.f - z) * zb) * o * tanhf(cn);
      p.c2[idx]  = cn;
      p.h2f[idx] = hn;
      p.h2[idx]  = (bf16)hn;
      p.h2z[idx] = (bf16)(zb * hn);          // z1_new * h2_new for next step
      if (j == 0) p.z2[wp * Bb + b] = zn;
    }
    grid_sync(p.bar);

    // ---- Phase 5: out_t = h1@l1_W^T + h2@l2_W^T + biases ----------------
    if (gw < 64) {                 // 16 M-tiles x 4 N-tiles of 16x16
      int m0 = (gw >> 2) << 4;
      int n0 = (gw & 3) << 4;
      v8f acc = {};
      for (int ks = 0; ks < KSLABS; ++ks) {
        int k0 = ks << 5;
        v16bf a = load_fragA(p.h1, Hh, m0, k0, lane);
        v16bf b = load_fragB(p.l1W, Hh, n0, k0, lane);
        acc = __builtin_amdgcn_wmma_f32_16x16x32_bf16(
            false, a, false, b, (short)0, acc, false, false);
      }
      for (int ks = 0; ks < KSLABS; ++ks) {
        int k0 = ks << 5;
        v16bf a = load_fragA(p.h2, Hh, m0, k0, lane);
        v16bf b = load_fragB(p.l2W, Hh, n0, k0, lane);
        acc = __builtin_amdgcn_wmma_f32_16x16x32_bf16(
            false, a, false, b, (short)0, acc, false, false);
      }
      int col   = n0 + (lane & 15);
      int rbase = m0 + ((lane >> 4) << 3);
      float bias = p.l1b[col] + p.l2b[col];
#pragma unroll
      for (int i = 0; i < 8; ++i)
        p.out[((size_t)(rbase + i) * Tt + t) * Dd + col] = acc[i] + bias;
    }
    // no barrier needed: next phase-1 only reads state, writes s1
  }
}

// ------------------------------ setup kernels -------------------------------
// fp32 [K][Norig] -> bf16 [Npad][K] (transposed, zero-padded columns)
__global__ void k_transpose_bf16(const float* in, bf16* out, int K, int Norig, int Npad) {
  int idx = blockIdx.x * blockDim.x + threadIdx.x;
  if (idx >= Npad * K) return;
  int n = idx / K, k = idx - n * K;
  float v = (n < Norig) ? in[(size_t)k * Norig + n] : 0.f;
  out[idx] = (bf16)v;
}

__global__ void k_convert_bf16(const float* in, bf16* out, int total) {
  int idx = blockIdx.x * blockDim.x + threadIdx.x;
  if (idx < total) out[idx] = (bf16)in[idx];
}

__global__ void k_init(bf16* h1, bf16* h2, bf16* h1z, bf16* h2z,
                       float* c1, float* c2, float* h2f,
                       float* z1, float* z2, unsigned* bar) {
  int idx = blockIdx.x * blockDim.x + threadIdx.x;
  if (idx < Bb * Hh) {
    h1[idx] = (bf16)0.f; h2[idx] = (bf16)0.f;
    h1z[idx] = (bf16)0.f; h2z[idx] = (bf16)0.f;
    c1[idx] = 0.f; c2[idx] = 0.f; h2f[idx] = 0.f;
  }
  if (idx < 2 * Bb) { z1[idx] = 0.f; z2[idx] = 0.f; }
  if (idx < 2) bar[idx] = 0u;
}

// ------------------------------- host launcher ------------------------------
extern "C" void kernel_launch(void* const* d_in, const int* in_sizes, int n_in,
                              void* d_out, int out_size, void* d_ws, size_t ws_size,
                              hipStream_t stream) {
  const float* x      = (const float*)d_in[0];
  const float* U11_1  = (const float*)d_in[1];
  const float* U21_1  = (const float*)d_in[2];
  const float* W01_1  = (const float*)d_in[3];
  const float* bias1  = (const float*)d_in[4];
  const float* U11_2  = (const float*)d_in[5];
  const float* W01_2  = (const float*)d_in[6];
  const float* bias2  = (const float*)d_in[7];
  const float* l1_W   = (const float*)d_in[8];
  const float* l1_b   = (const float*)d_in[9];
  const float* l2_W   = (const float*)d_in[10];
  const float* l2_b   = (const float*)d_in[11];

  char* ws = (char*)d_ws;
  size_t off = 0;
  auto alloc = [&](size_t bytes) -> char* {
    char* p = ws + off;
    off += (bytes + 255) & ~(size_t)255;
    return p;
  };
  bf16* U11_1t = (bf16*)alloc((size_t)NPAD * Hh * 2);
  bf16* U21_1t = (bf16*)alloc((size_t)NPAD * Hh * 2);
  bf16* U11_2t = (bf16*)alloc((size_t)NPAD * Hh * 2);
  bf16* W01_2t = (bf16*)alloc((size_t)NPAD * Hh * 2);
  bf16* W01_1t = (bf16*)alloc((size_t)NPAD * Dd * 2);
  bf16* l1W    = (bf16*)alloc((size_t)Dd * Hh * 2);
  bf16* l2W    = (bf16*)alloc((size_t)Dd * Hh * 2);
  float* s1    = (float*)alloc((size_t)Bb * NPAD * 4);
  float* s2    = (float*)alloc((size_t)Bb * NPAD * 4);
  float* c1    = (float*)alloc((size_t)Bb * Hh * 4);
  float* c2    = (float*)alloc((size_t)Bb * Hh * 4);
  float* h2f   = (float*)alloc((size_t)Bb * Hh * 4);
  bf16* h1     = (bf16*)alloc((size_t)Bb * Hh * 2);
  bf16* h2     = (bf16*)alloc((size_t)Bb * Hh * 2);
  bf16* h1z    = (bf16*)alloc((size_t)Bb * Hh * 2);
  bf16* h2z    = (bf16*)alloc((size_t)Bb * Hh * 2);
  float* z1    = (float*)alloc(2 * Bb * 4);
  float* z2    = (float*)alloc(2 * Bb * 4);
  unsigned* bar = (unsigned*)alloc(256);

  const int thr = 256;
  int nbig = NPAD * Hh;                 // transposed big-weight elements
  k_transpose_bf16<<<(nbig + thr - 1) / thr, thr, 0, stream>>>(U11_1, U11_1t, Hh, 4 * Hh + 1, NPAD);
  k_transpose_bf16<<<(nbig + thr - 1) / thr, thr, 0, stream>>>(U21_1, U21_1t, Hh, 4 * Hh + 1, NPAD);
  k_transpose_bf16<<<(nbig + thr - 1) / thr, thr, 0, stream>>>(U11_2, U11_2t, Hh, 4 * Hh + 1, NPAD);
  k_transpose_bf16<<<(nbig + thr - 1) / thr, thr, 0, stream>>>(W01_2, W01_2t, Hh, 4 * Hh + 1, NPAD);
  int nsm = NPAD * Dd;
  k_transpose_bf16<<<(nsm + thr - 1) / thr, thr, 0, stream>>>(W01_1, W01_1t, Dd, 4 * Hh + 1, NPAD);
  k_convert_bf16<<<(Dd * Hh + thr - 1) / thr, thr, 0, stream>>>(l1_W, l1W, Dd * Hh);
  k_convert_bf16<<<(Dd * Hh + thr - 1) / thr, thr, 0, stream>>>(l2_W, l2W, Dd * Hh);
  k_init<<<(Bb * Hh + thr - 1) / thr, thr, 0, stream>>>(h1, h2, h1z, h2z, c1, c2, h2f, z1, z2, bar);

  Params p;
  p.x = x; p.bias1 = bias1; p.bias2 = bias2; p.l1b = l1_b; p.l2b = l2_b;
  p.U11_1t = U11_1t; p.U21_1t = U21_1t; p.W01_1t = W01_1t;
  p.U11_2t = U11_2t; p.W01_2t = W01_2t; p.l1W = l1W; p.l2W = l2W;
  p.s1 = s1; p.s2 = s2; p.c1 = c1; p.c2 = c2; p.h2f = h2f;
  p.h1 = h1; p.h2 = h2; p.h1z = h1z; p.h2z = h2z;
  p.z1 = z1; p.z2 = z2; p.bar = bar; p.out = (float*)d_out;

  hmlstm_persistent<<<NBLK, NTHR, 0, stream>>>(p);
}
